// LSTMModel_30906584662124
// MI455X (gfx1250) — matrix operand order repaired and checked by reading, not compiled, not run on hardware
//
#include <hip/hip_runtime.h>

typedef __attribute__((ext_vector_type(16))) _Float16 v16h;
typedef __attribute__((ext_vector_type(8)))  float    v8f;
typedef __attribute__((ext_vector_type(4)))  unsigned int u32x4;
typedef __attribute__((ext_vector_type(8)))  int     i32x8;
typedef __attribute__((ext_vector_type(4)))  int     i32x4;

namespace {
constexpr int kH   = 64;    // hidden
constexpr int kF   = 34;    // input features
constexpr int kT   = 512;   // timesteps
constexpr int kBT  = 32;    // batch rows per workgroup (2 M-tiles)
constexpr int kThreads = 256;   // 8 wave32 waves

constexpr int kXStr = 68;   // x-stage row stride in floats (64 data + 4 pad) -> 0 bank conflicts
constexpr int kHStr = 72;   // h row stride in halves (64 data + 8 pad)      -> 0 bank conflicts

// LDS layout (bytes)
constexpr int kXBufBytes = kBT * kXStr * 4;            // 8704 per x buffer
constexpr int kOffBias = 4 * 2 * 16 * 32 * 16 * 2;     // weight frags: 131072
constexpr int kOffX    = kOffBias + 512 * 4;           // 133120 (x fp32, double buffered)
constexpr int kOffH0   = kOffX + 2 * kXBufBytes;       // 150528
constexpr int kOffH1   = kOffH0 + kBT * kHStr * 2;     // 155136
constexpr int kSmemBytes = kOffH1 + kBT * kHStr * 2;   // 159744 (<160KB -> 2 WG/WGP)
}

__device__ __forceinline__ float sigmoid_f(float x) {
  return 1.0f / (1.0f + __expf(-x));
}
__device__ __forceinline__ float tanh_f(float x) {
  return 1.0f - 2.0f / (__expf(2.0f * x) + 1.0f);
}

__device__ __forceinline__ v8f wmma16(v16h a, v16h b, v8f c) {
  return __builtin_amdgcn_wmma_f32_16x16x32_f16(
      /*neg_a=*/false, a, /*neg_b=*/false, b,
      /*c_mod=*/(short)0, c, /*reuse_a=*/false, /*reuse_b=*/false);
}

// ---- TDM: DMA one x tile (32 rows x 34 f32, row stride T*F) into LDS -------
// tile_dim0=64 > tensor_dim0=34  -> TDM zero-fills K padding (OOB reads = 0).
// pad_enable, interval=64 DW, amount=4 DW -> LDS row stride 68 floats.
__device__ __forceinline__ void tdm_load_xtile(const float* gsrc, unsigned ldsOff) {
  const unsigned long long ga = (unsigned long long)gsrc;
  u32x4 g0;
  g0[0] = 1u;                                              // count=1 (valid user D#)
  g0[1] = ldsOff;                                          // lds_addr (bytes)
  g0[2] = (unsigned)(ga & 0xffffffffu);                    // global_addr[31:0]
  g0[3] = (unsigned)((ga >> 32) & 0x1ffffffu) | (2u << 30);// global_addr[56:32] | type=2

  i32x8 g1;
  g1[0] = (int)((2u << 16)      // data_size = 4 bytes
              | (1u << 20)      // pad_enable
              | (5u << 22)      // pad_interval: 64 DWORDs
              | (3u << 25));    // pad_amount:   4 DWORDs
  g1[1] = (int)(34u << 16);     // tensor_dim0 = 34 (valid elems per row)
  g1[2] = (int)(32u << 16);     // tensor_dim0 hi=0 | tensor_dim1 = 32 rows
  g1[3] = (int)(64u << 16);     // tensor_dim1 hi=0 | tile_dim0 = 64
  g1[4] = 32;                   // tile_dim1 = 32 | tile_dim2 = 0
  g1[5] = 17408;                // tensor_dim0_stride = T*F = 512*34
  g1[6] = 0;                    // stride hi | tensor_dim1_stride lo
  g1[7] = 0;

  i32x4 gz = {0, 0, 0, 0};
#if __has_include(<hip/amd_detail/amd_gfx1250_TDM.h>)
  i32x8 gz8 = {0, 0, 0, 0, 0, 0, 0, 0};
  __builtin_amdgcn_tensor_load_to_lds(g0, g1, gz, gz, gz8, 0);   // clang-23 form
#else
  __builtin_amdgcn_tensor_load_to_lds(g0, g1, gz, gz, 0);        // ROCm 7.2 form
#endif
}

// A-fragment (16x32, ISA 7.12.2): lane = row M=lane%16; halves 0..7 hold
// K=(lane<16?0:8)+j, halves 8..15 hold K=(lane<16?16:24)+(j-8). Two contiguous
// runs -> ds_load_b128 pairs.
__device__ __forceinline__ v16h afrag_f16(const _Float16* buf, int mtile,
                                          int kt, int lane) {
  const _Float16* row = buf + (mtile * 16 + (lane & 15)) * kHStr + kt * 32;
  const int lo = (lane & 16) ? 8 : 0;
  const int hi = (lane & 16) ? 24 : 16;
  v16h a;
#pragma unroll
  for (int j = 0; j < 8; ++j) a[j] = row[lo + j];
#pragma unroll
  for (int j = 0; j < 8; ++j) a[8 + j] = row[hi + j];
  return a;
}
__device__ __forceinline__ v16h afrag_f32(const float* buf, int mtile,
                                          int kt, int lane) {
  const float* row = buf + (mtile * 16 + (lane & 15)) * kXStr + kt * 32;
  const int lo = (lane & 16) ? 8 : 0;
  const int hi = (lane & 16) ? 24 : 16;
  v16h a;
#pragma unroll
  for (int j = 0; j < 8; ++j) a[j] = (_Float16)row[lo + j];
#pragma unroll
  for (int j = 0; j < 8; ++j) a[8 + j] = (_Float16)row[hi + j];
  return a;
}

__device__ __forceinline__ v16h bfrag(const _Float16* wf, int w, int kt,
                                      int nt, int lane) {
  return *(const v16h*)(wf + ((((w * 2 + kt) * 16 + nt) * 32) + lane) * 16);
}

// One LSTM layer, one timestep, per wave: gates = aX*Wx^T + aH*Wh^T + bias,
// then elementwise update of c (kept in registers in C-fragment layout).
template <bool XF32>
__device__ __forceinline__ void lstm_step(const void* aX, const _Float16* aH,
                                          const _Float16* wf, int wXi, int wHi,
                                          const float* biasL, int mtile, int q,
                                          int lane, v8f& c, v8f& hOut) {
  const int ln = lane & 15;
  v8f acc0, acc1, acc2, acc3;
  const float b0 = biasL[  0 + (q << 4) + ln];
  const float b1 = biasL[ 64 + (q << 4) + ln];
  const float b2 = biasL[128 + (q << 4) + ln];
  const float b3 = biasL[192 + (q << 4) + ln];
#pragma unroll
  for (int r = 0; r < 8; ++r) { acc0[r] = b0; acc1[r] = b1; acc2[r] = b2; acc3[r] = b3; }

#pragma unroll
  for (int kt = 0; kt < 2; ++kt) {   // input contribution
    v16h a = XF32 ? afrag_f32((const float*)aX, mtile, kt, lane)
                  : afrag_f16((const _Float16*)aX, mtile, kt, lane);
    acc0 = wmma16(a, bfrag(wf, wXi, kt,  0 + q, lane), acc0);
    acc1 = wmma16(a, bfrag(wf, wXi, kt,  4 + q, lane), acc1);
    acc2 = wmma16(a, bfrag(wf, wXi, kt,  8 + q, lane), acc2);
    acc3 = wmma16(a, bfrag(wf, wXi, kt, 12 + q, lane), acc3);
  }
#pragma unroll
  for (int kt = 0; kt < 2; ++kt) {   // recurrent contribution
    v16h a = afrag_f16(aH, mtile, kt, lane);
    acc0 = wmma16(a, bfrag(wf, wHi, kt,  0 + q, lane), acc0);
    acc1 = wmma16(a, bfrag(wf, wHi, kt,  4 + q, lane), acc1);
    acc2 = wmma16(a, bfrag(wf, wHi, kt,  8 + q, lane), acc2);
    acc3 = wmma16(a, bfrag(wf, wHi, kt, 12 + q, lane), acc3);
  }

#pragma unroll
  for (int r = 0; r < 8; ++r) {
    const float iv = sigmoid_f(acc0[r]);
    const float fv = sigmoid_f(acc1[r]);
    const float gv = tanh_f(acc2[r]);
    const float ov = sigmoid_f(acc3[r]);
    const float cv = fv * c[r] + iv * gv;
    c[r] = cv;
    hOut[r] = ov * tanh_f(cv);
  }
}

__global__ void __launch_bounds__(kThreads)
lstm2_fused(const float* __restrict__ x,
            const float* __restrict__ Wih0, const float* __restrict__ Whh0,
            const float* __restrict__ bih0, const float* __restrict__ bhh0,
            const float* __restrict__ Wih1, const float* __restrict__ Whh1,
            const float* __restrict__ bih1, const float* __restrict__ bhh1,
            const float* __restrict__ Wfc,  const float* __restrict__ bfc,
            float* __restrict__ out) {
  extern __shared__ __attribute__((aligned(128))) char smem[];
  _Float16* wf    = (_Float16*)smem;
  float*    bias  = (float*)(smem + kOffBias);
  _Float16* h0buf = (_Float16*)(smem + kOffH0);
  _Float16* h1buf = (_Float16*)(smem + kOffH1);

  const int tid   = threadIdx.x;
  const int lane  = tid & 31;
  const int wave  = tid >> 5;
  const int mtile = wave >> 2;   // which 16 batch rows
  const int q     = wave & 3;    // which 16 hidden units (all 4 gates)
  const int bBase = blockIdx.x * kBT;

  // ---- one-time: stage weights into LDS in WMMA B-fragment order (f16) -----
  // B layout (ISA 7.12.2): lane%16 = N column; halves 0..7 -> K=(lane<16?0:8)+j, halves 8..15 -> K=(lane<16?16:24)+(j-8).
  const float* Ws[4] = {Wih0, Whh0, Wih1, Whh1};
  const int    Kr[4] = {kF, kH, kH, kH};
#pragma unroll
  for (int w = 0; w < 4; ++w) {
    const float* src = Ws[w];
    const int kr = Kr[w];
    for (int idx = tid; idx < 2 * 16 * 32 * 16; idx += kThreads) {
      const int tile = idx >> 9;        // kt*16 + nt
      const int rem  = idx & 511;       // lane*16 + j
      const int ln_  = rem >> 4;
      const int j    = rem & 15;
      const int kt   = tile >> 4;
      const int nt   = tile & 15;
      const int k = kt * 32 + ((j & 8) ? 16 : 0) + ((ln_ & 16) ? 8 : 0) + (j & 7);
      const int n = nt * 16 + (ln_ & 15);
      const float v = (k < kr) ? src[n * kr + k] : 0.0f;   // zero-pad K
      wf[(w * 32 + tile) * 512 + rem] = (_Float16)v;
    }
  }
  for (int idx = tid; idx < 256; idx += kThreads) {
    bias[idx]       = bih0[idx] + bhh0[idx];
    bias[256 + idx] = bih1[idx] + bhh1[idx];
  }
  for (int idx = tid; idx < kBT * kHStr; idx += kThreads) {
    h0buf[idx] = (_Float16)0.0f;
    h1buf[idx] = (_Float16)0.0f;
  }

  // kick off the TDM DMA for timestep 0 (wave 0 only; TDM ignores EXEC)
  if (tid < 32)
    tdm_load_xtile(x + (size_t)bBase * kT * kF, (unsigned)kOffX);

  v8f c0, c1;
#pragma unroll
  for (int r = 0; r < 8; ++r) { c0[r] = 0.0f; c1[r] = 0.0f; }

  for (int t = 0; t < kT; ++t) {
    if (tid < 32)
      __builtin_amdgcn_s_wait_tensorcnt(0);   // x(t) tile landed in LDS
    __syncthreads();  // publish x(t) + h(t-1) writes + (t==0) weight staging
    if (tid < 32 && t + 1 < kT)               // prefetch next tile via TDM
      tdm_load_xtile(x + ((size_t)bBase * kT + (t + 1)) * kF,
                     (unsigned)(kOffX + ((t + 1) & 1) * kXBufBytes));

    const float* xcur = (const float*)(smem + kOffX + (t & 1) * kXBufBytes);

    v8f h0new;
    lstm_step<true>(xcur, h0buf, wf, 0, 1, bias, mtile, q, lane, c0, h0new);
    __syncthreads();  // all waves done reading h0buf(t-1)
#pragma unroll
    for (int r = 0; r < 8; ++r) {   // C/D layout: M = r + 8*(lane>=16)
      const int m = mtile * 16 + r + ((lane >> 4) << 3);
      h0buf[m * kHStr + (q << 4) + (lane & 15)] = (_Float16)h0new[r];
    }
    __syncthreads();  // h0buf(t) ready for layer 1

    v8f h1new;
    lstm_step<false>(h0buf, h1buf, wf, 2, 3, bias + 256, mtile, q, lane, c1, h1new);
    __syncthreads();  // all waves done reading h1buf(t-1)
#pragma unroll
    for (int r = 0; r < 8; ++r) {
      const int m = mtile * 16 + r + ((lane >> 4) << 3);
      h1buf[m * kHStr + (q << 4) + (lane & 15)] = (_Float16)h1new[r];
    }
    // next iteration's first __syncthreads() orders these writes before reads
  }
  __syncthreads();

  // linear head on final h1: [32 x 64] @ [64 x 3]^T + b
  if (tid < kBT * 3) {
    const int r = tid / 3;
    const int k = tid % 3;
    float s = bfc[k];
#pragma unroll
    for (int j = 0; j < kH; ++j)
      s += (float)h1buf[r * kHStr + j] * Wfc[k * kH + j];
    out[(size_t)(bBase + r) * 3 + k] = s;
  }
}

extern "C" void kernel_launch(void* const* d_in, const int* in_sizes, int n_in,
                              void* d_out, int out_size, void* d_ws, size_t ws_size,
                              hipStream_t stream) {
  (void)n_in; (void)out_size; (void)d_ws; (void)ws_size;
  const float* x    = (const float*)d_in[0];
  const float* Wih0 = (const float*)d_in[1];
  const float* Whh0 = (const float*)d_in[2];
  const float* bih0 = (const float*)d_in[3];
  const float* bhh0 = (const float*)d_in[4];
  const float* Wih1 = (const float*)d_in[5];
  const float* Whh1 = (const float*)d_in[6];
  const float* bih1 = (const float*)d_in[7];
  const float* bhh1 = (const float*)d_in[8];
  const float* Wfc  = (const float*)d_in[9];
  const float* bfc  = (const float*)d_in[10];
  float* out = (float*)d_out;

  const int B = in_sizes[0] / (kT * kF);   // 4096
  dim3 grid((B + kBT - 1) / kBT);          // 128 workgroups
  lstm2_fused<<<grid, kThreads, kSmemBytes, stream>>>(
      x, Wih0, Whh0, bih0, bhh0, Wih1, Whh1, bih1, bhh1, Wfc, bfc, out);
}
